// MultiHeadDilatedSliddingWindowAttention_15839839388321
// MI455X (gfx1250) — matrix operand order repaired and checked by reading, not compiled, not run on hardware
//
#include <hip/hip_runtime.h>
#include <hip/hip_bf16.h>

// MI455X (gfx1250) multi-head sliding-window attention.
// - All matmuls: v_wmma_f32_16x16x32_bf16 (bf16 in, f32 accumulate).
// - GEMM A tiles staged via the Tensor Data Mover (tensor_load_to_lds + D#),
//   B tiles staged transposed in LDS so every fragment is built from b128 loads.
// - Full-row softmax over the zero-padded band handled analytically:
//     out_i = [ sum_band (exp(s-m)-exp(-m))*V_j + exp(-m)*sum_all V ] / Z_i
//     Z_i   = sum_band exp(s-m) + (N - band_count_i)*exp(-m),  m = max(0, band max)

typedef __bf16 bf16_t;
typedef __attribute__((ext_vector_type(16))) __bf16 v16bf;
typedef __attribute__((ext_vector_type(8)))  __bf16 v8bf;
typedef __attribute__((ext_vector_type(8)))  float  v8f;
typedef __attribute__((ext_vector_type(4)))  float  f32x4;
typedef __attribute__((ext_vector_type(4)))  unsigned int u32x4;
typedef __attribute__((ext_vector_type(4)))  int i32x4;
typedef __attribute__((ext_vector_type(8)))  int i32x8;

#define NSEQ 2048
#define EDIM 1024
#define NHEAD 16
#define DHEAD 64
#define HDIM 1024   // NHEAD * DHEAD

static __device__ __forceinline__ v16bf concat8(v8bf lo, v8bf hi) {
  return __builtin_shufflevector(lo, hi, 0, 1, 2, 3, 4, 5, 6, 7,
                                 8, 9, 10, 11, 12, 13, 14, 15);
}

// ---------------------------------------------------------------------------
// TDM: 2-D tile load Global -> LDS (bf16 elements).  D# per ISA 08 §8.
// ---------------------------------------------------------------------------
static __device__ __forceinline__ void tdm_load_2d_bf16(
    unsigned lds_off, const bf16_t* gptr,
    int tdim0, int tdim1, int stride0, int tile0, int tile1) {
#if __has_builtin(__builtin_amdgcn_tensor_load_to_lds)
  unsigned long long ga = (unsigned long long)gptr;
  u32x4 g0;
  g0[0] = 1u;                                   // count=1, user descriptor
  g0[1] = lds_off;                              // LDS byte address
  g0[2] = (unsigned)(ga & 0xffffffffu);         // global_addr[31:0]
  g0[3] = (unsigned)((ga >> 32) & 0x1ffffffu)   // global_addr[56:32]
        | (2u << 30);                           // type=2 ("image")
  i32x8 g1;
  g1[0] = (1 << 16);                            // data_size=1 (2B), mask=0
  g1[1] = (tdim0 & 0xffff) << 16;               // tensor_dim0[15:0]
  g1[2] = ((tdim0 >> 16) & 0xffff) | ((tdim1 & 0xffff) << 16);
  g1[3] = ((tdim1 >> 16) & 0xffff) | ((tile0 & 0xffff) << 16);
  g1[4] = (tile1 & 0xffff);                     // tile_dim1; tile_dim2=0
  g1[5] = stride0;                              // tensor_dim0_stride[31:0]
  g1[6] = 0;
  g1[7] = 0;
  i32x4 gz = {};
#if defined(__clang_major__) && (__clang_major__ >= 23)
  i32x8 gz8 = {};
  __builtin_amdgcn_tensor_load_to_lds(g0, g1, gz, gz, gz8, 0);
#else
  __builtin_amdgcn_tensor_load_to_lds(g0, g1, gz, gz, 0);
#endif
#endif
}

// ---------------------------------------------------------------------------
// fp32 -> bf16 conversion (8 elements / thread)
// ---------------------------------------------------------------------------
__global__ void cvt_bf16_kernel(const float* __restrict__ src,
                                bf16_t* __restrict__ dst, int n) {
  int i = (blockIdx.x * blockDim.x + threadIdx.x) * 8;
  if (i >= n) return;
  f32x4 a = *(const f32x4*)(src + i);
  f32x4 b = *(const f32x4*)(src + i + 4);
  v8bf o;
#pragma unroll
  for (int q = 0; q < 4; ++q) { o[q] = (bf16_t)a[q]; o[4 + q] = (bf16_t)b[q]; }
  *(v8bf*)(dst + i) = o;
}

// ---------------------------------------------------------------------------
// V column sums: Vsum[c] = sum_n V[n][c]
// ---------------------------------------------------------------------------
__global__ void vsum_kernel(const bf16_t* __restrict__ V,
                            float* __restrict__ Vsum) {
  int c = blockIdx.x * blockDim.x + threadIdx.x;
  if (c >= HDIM) return;
  float s = 0.f;
  for (int n = 0; n < NSEQ; ++n) s += (float)V[(size_t)n * HDIM + c];
  Vsum[c] = s;
}

// ---------------------------------------------------------------------------
// GEMM: C[M x Nout] = A[M x K](bf16) @ W[K x Nout](bf16) + bias
// 256 threads = 8 waves (2x4); block tile 32(M) x 64(N), K staged in 32-steps.
// A tile: TDM (tensor_load_to_lds), row-major [32][32].
// B tile: manual transposed staging Bst[n][k] so fragments are contiguous.
// ---------------------------------------------------------------------------
__global__ __launch_bounds__(256)
void gemm_bf16_kernel(const bf16_t* __restrict__ A,
                      const bf16_t* __restrict__ W,
                      const float* __restrict__ bias,
                      bf16_t* __restrict__ Cb,       // bf16 output (or null)
                      float*  __restrict__ Cf,       // f32 output (or null)
                      int M, int K, int Nout) {
  __shared__ bf16_t As[32][32];     // row-major (TDM fills linearly)
  __shared__ bf16_t Bst[64][40];    // transposed: [n][k], 80B row stride

  const int t    = threadIdx.x;
  const int wave = t >> 5;
  const int lane = t & 31;
  const int wr   = wave >> 2;
  const int wc   = wave & 3;
  const int row0 = blockIdx.y * 32;
  const int col0 = blockIdx.x * 64;
  const int m    = lane & 15;
  const int half = lane >> 4;
  const unsigned as_off = (unsigned)(uintptr_t)(&As[0][0]);

  v8f acc = {};

  for (int kk = 0; kk < K; kk += 32) {
    // ---- A tile via Tensor Data Mover (wave 0 issues, all waves barrier) ----
#if __has_builtin(__builtin_amdgcn_tensor_load_to_lds)
    if (t < 32) {
      tdm_load_2d_bf16(as_off, A + (size_t)row0 * K + kk,
                       K, M, K, 32, 32);
    }
#else
    {
      int r = t >> 3, c4 = (t & 7) * 4;
      const bf16_t* src = A + (size_t)(row0 + r) * K + kk + c4;
#pragma unroll
      for (int q = 0; q < 4; ++q) As[r][c4 + q] = src[q];
    }
#endif
    // ---- B tile 32x64, stored transposed (coalesced read, b16 scatter) ----
    {
      int r = t >> 3, c8 = (t & 7) * 8;
      v8bf wv = *(const v8bf*)(W + (size_t)(kk + r) * Nout + col0 + c8);
#pragma unroll
      for (int q = 0; q < 8; ++q) Bst[c8 + q][r] = wv[q];
      if (kk + 32 < K)
        __builtin_prefetch(W + (size_t)(kk + 32 + r) * Nout + col0 + c8, 0, 1);
    }
#if __has_builtin(__builtin_amdgcn_tensor_load_to_lds)
    if (t < 32) __builtin_amdgcn_s_wait_tensorcnt(0);
#endif
    __syncthreads();

    // ---- fragments: all b128 LDS loads ----
    const bf16_t* arow = &As[wr * 16 + m][8 * half];
    v16bf a = concat8(*(const v8bf*)arow, *(const v8bf*)(arow + 16));
    const bf16_t* brow = &Bst[wc * 16 + m][8 * half];
    v16bf b = concat8(*(const v8bf*)brow, *(const v8bf*)(brow + 16));

    acc = __builtin_amdgcn_wmma_f32_16x16x32_bf16(false, a, false, b,
                                                  (short)0, acc, false, false);
    __syncthreads();
  }

  // ---- epilogue (C layout: lane=col, VGPR r -> row r + 8*half) ----
  const int ccol = col0 + wc * 16 + m;
  const float bv = bias ? bias[ccol] : 0.f;
#pragma unroll
  for (int r = 0; r < 8; ++r) {
    int crow = row0 + wr * 16 + r + 8 * half;
    float v = acc[r] + bv;
    if (Cf) Cf[(size_t)crow * Nout + ccol] = v;
    else    Cb[(size_t)crow * Nout + ccol] = (bf16_t)v;
  }
}

// ---------------------------------------------------------------------------
// Band attention: one wave per (head, 16-row query tile).
// Keys j in [i0-64, i0+95] -> 160 columns = 10 sub-tiles of 16.
// ---------------------------------------------------------------------------
__global__ __launch_bounds__(32)
void attn_kernel(const bf16_t* __restrict__ Q,
                 const bf16_t* __restrict__ Kmat,
                 const bf16_t* __restrict__ V,
                 const float*  __restrict__ Vsum,
                 bf16_t* __restrict__ AO) {
  __shared__ bf16_t Ps[16][176];    // shifted probs, 352B row stride (16B mult)

  const int h     = blockIdx.y;
  const int i0    = blockIdx.x * 16;
  const int lane  = threadIdx.x;
  const int m     = lane & 15;
  const int half  = lane >> 4;
  const int jbase = i0 - 64;
  const size_t hoff = (size_t)h * DHEAD;
  const v8bf zero8 = {};

  // ---- Q A-fragments: 2x b128 each ----
  const bf16_t* qrow = Q + (size_t)(i0 + m) * HDIM + hoff + 8 * half;
  v16bf aq0 = concat8(*(const v8bf*)qrow,        *(const v8bf*)(qrow + 16));
  v16bf aq1 = concat8(*(const v8bf*)(qrow + 32), *(const v8bf*)(qrow + 48));

  // ---- Scores: 10 tiles x (2 WMMAs over d=64); K^T B-frags straight from
  //      global K rows (d contiguous): 2x b128 per frag ----
  v8f sc[10];
#pragma unroll
  for (int tt = 0; tt < 10; ++tt) {
    int j = jbase + tt * 16 + m;
    bool jok = (j >= 0) && (j < NSEQ);
    const bf16_t* krow = Kmat + (size_t)(jok ? j : 0) * HDIM + hoff + 8 * half;
    v8bf k0 = jok ? *(const v8bf*)krow        : zero8;
    v8bf k1 = jok ? *(const v8bf*)(krow + 16) : zero8;
    v8bf k2 = jok ? *(const v8bf*)(krow + 32) : zero8;
    v8bf k3 = jok ? *(const v8bf*)(krow + 48) : zero8;
    v8f c = {};
    c = __builtin_amdgcn_wmma_f32_16x16x32_bf16(false, aq0, false,
                                                concat8(k0, k1),
                                                (short)0, c, false, false);
    c = __builtin_amdgcn_wmma_f32_16x16x32_bf16(false, aq1, false,
                                                concat8(k2, k3),
                                                (short)0, c, false, false);
    sc[tt] = c;
  }

  // ---- Row statistics (rows r + 8*half live in this lane-half) ----
  float mrow[8], zrow[8], cexp[8];
#pragma unroll
  for (int r = 0; r < 8; ++r) {
    const int rp = r + 8 * half;
    float mx = -3.4e38f;
#pragma unroll
    for (int tt = 0; tt < 10; ++tt) {
      int c = tt * 16 + m;
      int j = jbase + c;
      bool band = (c >= rp) && (c <= rp + 128) && (j >= 0) && (j < NSEQ);
      if (band) mx = fmaxf(mx, sc[tt][r]);
    }
#pragma unroll
    for (int off = 1; off < 16; off <<= 1)
      mx = fmaxf(mx, __shfl_xor(mx, off, 16));
    mx = fmaxf(mx, 0.f);                  // zeros outside the band join the max
    float se = 0.f;
#pragma unroll
    for (int tt = 0; tt < 10; ++tt) {
      int c = tt * 16 + m;
      int j = jbase + c;
      bool band = (c >= rp) && (c <= rp + 128) && (j >= 0) && (j < NSEQ);
      if (band) se += __expf(sc[tt][r] - mx);
    }
#pragma unroll
    for (int off = 1; off < 16; off <<= 1)
      se += __shfl_xor(se, off, 16);
    int i  = i0 + rp;
    int lo = (i - 64 < 0) ? 0 : i - 64;
    int hi = (i + 64 > NSEQ - 1) ? NSEQ - 1 : i + 64;
    int cnt = hi - lo + 1;
    float ce = __expf(-mx);
    mrow[r] = mx;
    cexp[r] = ce;
    zrow[r] = se + (float)(NSEQ - cnt) * ce;
  }

  // ---- P' = exp(s-m) - exp(-m) inside band, else 0 -> LDS ----
#pragma unroll
  for (int r = 0; r < 8; ++r) {
    const int rp = r + 8 * half;
#pragma unroll
    for (int tt = 0; tt < 10; ++tt) {
      int c = tt * 16 + m;
      int j = jbase + c;
      bool band = (c >= rp) && (c <= rp + 128) && (j >= 0) && (j < NSEQ);
      float p = band ? (__expf(sc[tt][r] - mrow[r]) - cexp[r]) : 0.f;
      Ps[rp][c] = (bf16_t)p;
    }
  }
  __syncthreads();

  // ---- P' @ V : 5 K-steps of 32 keys x 4 d-tiles of 16 ----
  v8f ao[4];
  {
    v8f z = {};
#pragma unroll
    for (int nt = 0; nt < 4; ++nt) ao[nt] = z;
  }
#pragma unroll
  for (int s = 0; s < 5; ++s) {
    const bf16_t* prow = &Ps[m][s * 32 + 8 * half];
    v16bf ap = concat8(*(const v8bf*)prow, *(const v8bf*)(prow + 16));
#pragma unroll
    for (int nt = 0; nt < 4; ++nt) {
      v16bf bv;
#pragma unroll
      for (int e = 0; e < 16; ++e) {              // keys strided: gather
        int k = (e & 7) + (half << 3) + ((e >> 3) << 4);
        int j = jbase + s * 32 + k;
        bool jok = (j >= 0) && (j < NSEQ);
        bv[e] = jok ? V[(size_t)j * HDIM + hoff + nt * 16 + m] : (bf16_t)0.f;
      }
      ao[nt] = __builtin_amdgcn_wmma_f32_16x16x32_bf16(false, ap, false, bv,
                                                       (short)0, ao[nt],
                                                       false, false);
    }
  }

  // ---- Epilogue: add c_i * sum(V), normalize by Z ----
#pragma unroll
  for (int nt = 0; nt < 4; ++nt) {
    int d = nt * 16 + m;
    float vs = Vsum[h * DHEAD + d];
#pragma unroll
    for (int r = 0; r < 8; ++r) {
      int rp = r + 8 * half;
      float o = (ao[nt][r] + cexp[r] * vs) / zrow[r];
      AO[(size_t)(i0 + rp) * HDIM + hoff + d] = (bf16_t)o;
    }
  }
}

// ---------------------------------------------------------------------------
// Launch
// ---------------------------------------------------------------------------
extern "C" void kernel_launch(void* const* d_in, const int* in_sizes, int n_in,
                              void* d_out, int out_size, void* d_ws, size_t ws_size,
                              hipStream_t stream) {
  const float* x  = (const float*)d_in[0];
  const float* Wq = (const float*)d_in[1];
  const float* bq = (const float*)d_in[2];
  const float* Wk = (const float*)d_in[3];
  const float* bk = (const float*)d_in[4];
  const float* Wv = (const float*)d_in[5];
  const float* bv = (const float*)d_in[6];
  const float* Wo = (const float*)d_in[7];
  const float* bo = (const float*)d_in[8];
  float* out = (float*)d_out;

  bf16_t* xb  = (bf16_t*)d_ws;
  bf16_t* Wqb = xb  + (size_t)NSEQ * EDIM;
  bf16_t* Wkb = Wqb + (size_t)EDIM * HDIM;
  bf16_t* Wvb = Wkb + (size_t)EDIM * HDIM;
  bf16_t* Wob = Wvb + (size_t)EDIM * HDIM;
  bf16_t* Qb  = Wob + (size_t)EDIM * HDIM;
  bf16_t* Kb  = Qb  + (size_t)NSEQ * HDIM;
  bf16_t* Vb  = Kb  + (size_t)NSEQ * HDIM;
  bf16_t* AOb = Vb  + (size_t)NSEQ * HDIM;
  float*  Vsum = (float*)(AOb + (size_t)NSEQ * HDIM);

  const int nx = NSEQ * EDIM;       // 2M elements
  const int nw = EDIM * HDIM;       // 1M elements
  cvt_bf16_kernel<<<nx / (256 * 8), 256, 0, stream>>>(x,  xb,  nx);
  cvt_bf16_kernel<<<nw / (256 * 8), 256, 0, stream>>>(Wq, Wqb, nw);
  cvt_bf16_kernel<<<nw / (256 * 8), 256, 0, stream>>>(Wk, Wkb, nw);
  cvt_bf16_kernel<<<nw / (256 * 8), 256, 0, stream>>>(Wv, Wvb, nw);
  cvt_bf16_kernel<<<nw / (256 * 8), 256, 0, stream>>>(Wo, Wob, nw);

  dim3 gg(HDIM / 64, NSEQ / 32);    // (16, 64)
  gemm_bf16_kernel<<<gg, 256, 0, stream>>>(xb, Wqb, bq, Qb, nullptr,
                                           NSEQ, EDIM, HDIM);
  gemm_bf16_kernel<<<gg, 256, 0, stream>>>(xb, Wkb, bk, Kb, nullptr,
                                           NSEQ, EDIM, HDIM);
  gemm_bf16_kernel<<<gg, 256, 0, stream>>>(xb, Wvb, bv, Vb, nullptr,
                                           NSEQ, EDIM, HDIM);

  vsum_kernel<<<HDIM / 256, 256, 0, stream>>>(Vb, Vsum);

  attn_kernel<<<dim3(NSEQ / 16, NHEAD), 32, 0, stream>>>(Qb, Kb, Vb, Vsum, AOb);

  gemm_bf16_kernel<<<gg, 256, 0, stream>>>(AOb, Wob, bo, nullptr, out,
                                           NSEQ, HDIM, EDIM);
}
